// DMM_8358006358025
// MI455X (gfx1250) — matrix operand order repaired and checked by reading, not compiled, not run on hardware
//
#include <hip/hip_runtime.h>

// Problem dims
#define Tn 1024
#define Bn 64
#define Yn 512
#define Xn 128
#define En 512
#define Rn 512

// d_out float offsets (q_x,q_mu,q_lv,p_x,p_mu,p_lv,y_probs)
#define QX  ((size_t)0)
#define QMU ((size_t)8388608)
#define QLV ((size_t)16777216)
#define PX  ((size_t)25165824)
#define PMU ((size_t)33554432)
#define PLV ((size_t)41943040)
#define YP  ((size_t)50331648)

typedef __attribute__((ext_vector_type(16))) __bf16 v16bf;
typedef __attribute__((ext_vector_type(8)))  float  v8f;
typedef __attribute__((ext_vector_type(8)))  unsigned short u16x8;
typedef __attribute__((ext_vector_type(4)))  unsigned short u16x4;
typedef __attribute__((ext_vector_type(4)))  float  f32x4;

union BFV { v16bf v; u16x8 u[2]; unsigned short h[16]; };

__device__ __forceinline__ unsigned short f2bf(float f) {
  unsigned u = __float_as_uint(f);
  return (unsigned short)((u + 0x7FFFu + ((u >> 16) & 1u)) >> 16);  // RNE
}
__device__ __forceinline__ float bf2f(unsigned short h) {
  return __uint_as_float(((unsigned)h) << 16);
}
__device__ __forceinline__ v8f wmma_bf16(v16bf a, v16bf b, v8f c) {
  // v_wmma_f32_16x16x32_bf16 (wave32, f32 accum)
  return __builtin_amdgcn_wmma_f32_16x16x32_bf16(false, a, false, b, (short)0, c,
                                                 false, false);
}

// A fragment 16x32 bf16 from LDS (row-major 16 x lda tile), k offset kb.
// ISA 7.12.2: lane m=lane&15; regs 0-3 hold K=khi..khi+7, regs 4-7 K=16+khi..23+khi
// (khi = 8 for lanes 16-31) -> two contiguous 16-byte runs.
__device__ __forceinline__ v16bf load_a_lds(const unsigned short* at, int lda, int kb) {
  int lane = threadIdx.x & 31;
  const unsigned short* p = at + (lane & 15) * lda + kb + ((lane >> 4) << 3);
  BFV f;
  f.u[0] = *(const u16x8*)(p);
  f.u[1] = *(const u16x8*)(p + 16);
  return f.v;
}
// B fragment 32x16 from PRE-TRANSPOSED bf16 weights Wt (N x K row-major):
// col n = lane&15, K = kb + (lane>=16 ? 8 : 0) + e for e<8, +16 for e>=8 (matches A).
__device__ __forceinline__ v16bf load_b_t(const unsigned short* __restrict__ Wt,
                                          int Kstride, int kb, int nb) {
  int lane = threadIdx.x & 31;
  const unsigned short* p =
      Wt + (size_t)(nb + (lane & 15)) * Kstride + kb + ((lane >> 4) << 3);
  BFV f;
  f.u[0] = *(const u16x8*)(p);
  f.u[1] = *(const u16x8*)(p + 16);
  return f.v;
}

// 64-row block GEMM: stages A (64 x K, fp32 or bf16, row stride a_rs) into LDS as
// bf16 in 64x256 chunks; waves accumulate up to 4 16x16 tiles. A-fragment is
// hoisted across tiles that share the same row-block. Weights are bf16 N x K.
// Must be called uniformly by the whole workgroup (contains __syncthreads).
#define KC 256
__device__ __forceinline__ void mm64(const void* A, int a_bf16, size_t a_rs, int K,
                                     const unsigned short* const* Ws,
                                     int ntile, const int* mts, const int* nbs,
                                     v8f* acc, unsigned short* smem) {
  v8f z = {};
  for (int i = 0; i < 4; ++i) acc[i] = z;
  for (int kb0 = 0; kb0 < K; kb0 += KC) {
    int kc = (K - kb0 < KC) ? (K - kb0) : KC;   // 128 or 256 here
    int lg = (kc == 256) ? 8 : 7;
    __syncthreads();
    for (int i = threadIdx.x * 4; i < (64 << lg); i += blockDim.x * 4) {
      int r = i >> lg, c = i & (kc - 1);
      u16x4 o;
      if (a_bf16) {
        o = *(const u16x4*)((const unsigned short*)A + (size_t)r * a_rs + kb0 + c);
      } else {
        f32x4 w = *(const f32x4*)((const float*)A + (size_t)r * a_rs + kb0 + c);
        o[0] = f2bf(w[0]); o[1] = f2bf(w[1]); o[2] = f2bf(w[2]); o[3] = f2bf(w[3]);
      }
      *(u16x4*)(smem + i) = o;
    }
    __syncthreads();
    for (int kb = 0; kb < kc; kb += 32) {
      v16bf a = load_a_lds(smem + mts[0] * 16 * kc, kc, kb);
      int lastm = mts[0];
      for (int t = 0; t < ntile; ++t) {
        if (mts[t] != lastm) {
          a = load_a_lds(smem + mts[t] * 16 * kc, kc, kb);
          lastm = mts[t];
        }
        acc[t] = wmma_bf16(a, load_b_t(Ws[t], K, kb0 + kb, nbs[t]), acc[t]);
      }
    }
  }
}

// Device-scope barrier across nwg co-resident workgroups (partition-local counters).
__device__ __forceinline__ void grid_barrier(unsigned* cnt, unsigned* gen, unsigned nwg) {
  __threadfence();
  __syncthreads();
  if (threadIdx.x == 0) {
    unsigned g = atomicAdd(gen, 0u);
    if (atomicAdd(cnt, 1u) == nwg - 1u) {
      atomicExch(cnt, 0u);
      atomicAdd(gen, 1u);
    } else {
      while (atomicAdd(gen, 0u) == g) __builtin_amdgcn_s_sleep(1);
    }
  }
  __syncthreads();
  __threadfence();
}

// ---------------------------------------------------------------------------
// Transpose + convert weights: Wt[n*K + k] = bf16(W[k*N + n]), N = 1<<lgN.
__global__ void wt_kernel(const float* __restrict__ W, unsigned short* __restrict__ Wt,
                          int KN, int lgN, int K) {
  int i = blockIdx.x * blockDim.x + threadIdx.x;
  if (i < KN) {
    int k = i >> lgN, n = i & ((1 << lgN) - 1);
    Wt[(size_t)n * K + k] = f2bf(W[i]);
  }
}

__global__ void init_kernel(float* h_buf, float* xq_buf, float* xg_buf, unsigned* bar,
                            const float* h0, const float* xq0, const float* x0) {
  int tid = threadIdx.x;
  for (int i = tid; i < Bn * Rn; i += blockDim.x) h_buf[i] = h0[i & (Rn - 1)];
  for (int i = tid; i < Bn * Xn; i += blockDim.x) {
    xq_buf[i] = xq0[i & (Xn - 1)];
    xg_buf[i] = x0[i & (Xn - 1)];
  }
  if (tid < 4) bar[tid] = 0u;
}

// x_proj[t][b][r] = data[b][T-1-t][:] . Wih[:,r] + bih[r] + bhh[r]
__global__ void __launch_bounds__(256)
xproj_kernel(const float* __restrict__ data, const unsigned short* __restrict__ WihT,
             const float* __restrict__ bih, const float* __restrict__ bhh,
             float* __restrict__ xproj) {
  __shared__ unsigned short smem[64 * KC];
  int t = blockIdx.x;
  int tsrc = Tn - 1 - t;
  int wave = threadIdx.x >> 5, lane = threadIdx.x & 31;
  int hi = (lane >> 4) << 3, lc = lane & 15;
  const float* Abase = data + (size_t)tsrc * Yn;  // row b strided by Tn*Yn
  int mts[4], nbs[4]; const unsigned short* Ws[4]; v8f acc[4];
  for (int j = 0; j < 2; ++j) {
    int sub = wave * 2 + j;                        // 16 subtiles of 64x64 block
    mts[j] = sub >> 2;
    nbs[j] = blockIdx.y * 64 + (sub & 3) * 16;
    Ws[j] = WihT;
  }
  mm64(Abase, 0, (size_t)Tn * Yn, Yn, Ws, 2, mts, nbs, acc, smem);
  for (int j = 0; j < 2; ++j) {
#pragma unroll
    for (int i = 0; i < 8; ++i) {
      int b = mts[j] * 16 + i + hi, r = nbs[j] + lc;
      xproj[((size_t)t * Bn + b) * Rn + r] = acc[j][i] + bih[r] + bhh[r];
    }
  }
}

// Generic fused GEMM: C = act(A @ W + bias), A is M x K (fp32/bf16), Wt bf16 NxK.
__global__ void __launch_bounds__(256)
gemm_act(const void* __restrict__ A, int a_bf16,
         const unsigned short* __restrict__ Wt, const float* __restrict__ bias,
         void* __restrict__ C, int c_bf16, int K, int N, int act) {
  __shared__ unsigned short smem[64 * KC];
  int wave = threadIdx.x >> 5, lane = threadIdx.x & 31;
  int hi = (lane >> 4) << 3, lc = lane & 15;
  size_t rowbase = (size_t)blockIdx.x * 64;
  const void* Arow = a_bf16 ? (const void*)(((const unsigned short*)A) + rowbase * K)
                            : (const void*)(((const float*)A) + rowbase * K);
  int mts[4], nbs[4]; const unsigned short* Ws[4]; v8f acc[4];
  for (int j = 0; j < 2; ++j) {
    int sub = wave * 2 + j;
    mts[j] = sub >> 2;
    nbs[j] = blockIdx.y * 64 + (sub & 3) * 16;
    Ws[j] = Wt;
  }
  mm64(Arow, a_bf16, K, K, Ws, 2, mts, nbs, acc, smem);
  for (int j = 0; j < 2; ++j) {
#pragma unroll
    for (int i = 0; i < 8; ++i) {
      int r = mts[j] * 16 + i + hi, n = nbs[j] + lc;
      float v = acc[j][i] + bias[n];
      if (act == 1) v = fmaxf(v, 0.0f);
      else if (act == 2) v = 1.0f / (1.0f + expf(-v));
      size_t m = rowbase + r;
      if (c_bf16) ((unsigned short*)C)[m * N + n] = f2bf(v);
      else        ((float*)C)[m * N + n] = v;
    }
  }
}

// Persistent scan kernel. WGs 0-7: backward RNN then inference scan.
// WGs 8-15: generative scan (independent, runs concurrently).
__global__ void __launch_bounds__(256)
scan_kernel(const float* __restrict__ xproj, const unsigned short* __restrict__ WhhT,
            unsigned short* __restrict__ rnnout,
            const unsigned short* __restrict__ WxhT, const float* __restrict__ bxh,
            const unsigned short* __restrict__ WhmT, const float* __restrict__ bhm,
            const unsigned short* __restrict__ WhlT, const float* __restrict__ bhl,
            const float* __restrict__ eps_q,
            const unsigned short* __restrict__ Wg1T, const float* __restrict__ bg1,
            const unsigned short* __restrict__ Wg2T, const float* __restrict__ bg2,
            const unsigned short* __restrict__ Wp1T, const float* __restrict__ bp1,
            const unsigned short* __restrict__ Wp2T, const float* __restrict__ bp2,
            const unsigned short* __restrict__ WmuT, const float* __restrict__ bmu,
            const unsigned short* __restrict__ WlvT, const float* __restrict__ blv,
            const float* __restrict__ eps_p, float* __restrict__ out,
            float* h_buf, float* hc_buf, float* xq_buf, float* xg_buf,
            float* g1_buf, float* p1_buf, float* pmrelu, unsigned* bar) {
  __shared__ unsigned short smem[64 * KC];
  __shared__ float sred[3 * 1024];
  const int wave = threadIdx.x >> 5, lane = threadIdx.x & 31;
  const int hi = (lane >> 4) << 3, lc = lane & 15;
  int mts[4], nbs[4]; const unsigned short* Ws[4]; v8f acc[4];

  if (blockIdx.x < 8) {
    unsigned *cnt = bar + 0, *gen = bar + 1;
    const int wid = blockIdx.x * 8 + wave;   // 0..63 waves in partition
    const int cb = blockIdx.x * 16;          // owned X-column slice

    // ---- backward RNN: h = relu(xp[t] + h @ Whh), rnn_out[T-1-t] = h ----
    for (int t = 0; t < Tn; ++t) {
      const float* hsrc = h_buf + (t & 1) * (Bn * Rn);
      float* hdst = h_buf + ((t + 1) & 1) * (Bn * Rn);
      for (int j = 0; j < 2; ++j) {
        int tile = wid * 2 + j;              // 4 x 32 tile grid
        mts[j] = tile >> 5; nbs[j] = (tile & 31) * 16; Ws[j] = WhhT;
      }
      mm64(hsrc, 0, Rn, Rn, Ws, 2, mts, nbs, acc, smem);
      for (int j = 0; j < 2; ++j) {
#pragma unroll
        for (int i = 0; i < 8; ++i) {
          int b = mts[j] * 16 + i + hi, r = nbs[j] + lc;
          float v = fmaxf(acc[j][i] + xproj[((size_t)t * Bn + b) * Rn + r], 0.0f);
          hdst[b * Rn + r] = v;
          rnnout[((size_t)(Tn - 1 - t) * Bn + b) * Rn + r] = f2bf(v);
        }
      }
      grid_barrier(cnt, gen, 8);
    }

    // ---- inference scan ----
    for (int t = 0; t < Tn; ++t) {
      const float* xprev = xq_buf + (t & 1) * (Bn * Xn);
      // stage1: h_c = 0.5*(tanh(x@Wxh + bxh) + rnn_out[t])
      for (int j = 0; j < 2; ++j) {
        int tile = wid * 2 + j;
        mts[j] = tile >> 5; nbs[j] = (tile & 31) * 16; Ws[j] = WxhT;
      }
      mm64(xprev, 0, Xn, Xn, Ws, 2, mts, nbs, acc, smem);
      for (int j = 0; j < 2; ++j) {
#pragma unroll
        for (int i = 0; i < 8; ++i) {
          int b = mts[j] * 16 + i + hi, r = nbs[j] + lc;
          float hr = bf2f(rnnout[((size_t)t * Bn + b) * Rn + r]);
          hc_buf[b * Rn + r] = 0.5f * (tanhf(acc[j][i] + bxh[r]) + hr);
        }
      }
      grid_barrier(cnt, gen, 8);
      // stage2: mu (waves 0-3) / lv (waves 4-7) for cols [cb, cb+16)
      mts[0] = wave & 3; nbs[0] = cb; Ws[0] = (wave < 4) ? WhmT : WhlT;
      mm64(hc_buf, 0, Rn, Rn, Ws, 1, mts, nbs, acc, smem);
      {
        const float* bsel = (wave < 4) ? bhm : bhl;
        float* osel = out + ((wave < 4) ? QMU : QLV);
        int slot = (wave < 4) ? 0 : 1024;
#pragma unroll
        for (int i = 0; i < 8; ++i) {
          int b = mts[0] * 16 + i + hi, x = cb + lc;
          float v = acc[0][i] + bsel[x];
          osel[(size_t)b * (Tn * Xn) + t * Xn + x] = v;
          sred[slot + b * 16 + lc] = v;
        }
      }
      __syncthreads();
      // stage3: x_t = mu + eps*exp(0.5*lv)
      for (int i = threadIdx.x; i < 1024; i += blockDim.x) {
        int b = i >> 4, x = cb + (i & 15);
        float xt = sred[i] + eps_q[((size_t)t * Bn + b) * Xn + x] * expf(0.5f * sred[1024 + i]);
        out[QX + (size_t)b * (Tn * Xn) + t * Xn + x] = xt;
        xq_buf[((t + 1) & 1) * (Bn * Xn) + b * Xn + x] = xt;
      }
      grid_barrier(cnt, gen, 8);
    }
  } else {
    unsigned *cnt = bar + 2, *gen = bar + 3;
    const int wg = blockIdx.x - 8;
    const int wid = wg * 8 + wave;
    const int cb = wg * 16;

    // ---- generative scan ----
    for (int t = 0; t < Tn; ++t) {
      const float* xprev = xg_buf + (t & 1) * (Bn * Xn);
      // stage1: g1 = relu(x@Wg1+bg1), p1 = relu(x@Wp1+bp1): 64x1024 combined
      for (int j = 0; j < 4; ++j) {
        int tile = wid * 4 + j;                       // 4 x 64 tile grid
        mts[j] = tile >> 6;
        int nt2 = tile & 63;
        nbs[j] = (nt2 & 31) * 16;
        Ws[j] = (nt2 < 32) ? Wg1T : Wp1T;
      }
      mm64(xprev, 0, Xn, Xn, Ws, 4, mts, nbs, acc, smem);
      for (int j = 0; j < 4; ++j) {
        int nt2 = (wid * 4 + j) & 63;
        float* buf = (nt2 < 32) ? g1_buf : p1_buf;
        const float* bb = (nt2 < 32) ? bg1 : bp1;
#pragma unroll
        for (int i = 0; i < 8; ++i) {
          int b = mts[j] * 16 + i + hi, r = nbs[j] + lc;
          buf[b * 512 + r] = fmaxf(acc[j][i] + bb[r], 0.0f);
        }
      }
      grid_barrier(cnt, gen, 8);
      // stage2a: gate = sigmoid(g1@Wg2 + bg2) for owned cols
      mts[0] = wave & 3; nbs[0] = cb;
      Ws[0] = Wg2T;
      mm64(g1_buf, 0, 512, 512, Ws, (wave < 4) ? 1 : 0, mts, nbs, acc, smem);
      if (wave < 4) {
#pragma unroll
        for (int i = 0; i < 8; ++i) {
          int b = mts[0] * 16 + i + hi;
          sred[b * 16 + lc] = 1.0f / (1.0f + expf(-(acc[0][i] + bg2[cb + lc])));
        }
      }
      // stage2b: pm = p1@Wp2 + bp2
      Ws[0] = Wp2T;
      mm64(p1_buf, 0, 512, 512, Ws, (wave < 4) ? 1 : 0, mts, nbs, acc, smem);
      if (wave < 4) {
#pragma unroll
        for (int i = 0; i < 8; ++i) {
          int b = mts[0] * 16 + i + hi;
          sred[1024 + b * 16 + lc] = acc[0][i] + bp2[cb + lc];
        }
      }
      // stage2c: xmu = x@Wmu + bmu
      Ws[0] = WmuT;
      mm64(xprev, 0, Xn, Xn, Ws, (wave < 4) ? 1 : 0, mts, nbs, acc, smem);
      if (wave < 4) {
#pragma unroll
        for (int i = 0; i < 8; ++i) {
          int b = mts[0] * 16 + i + hi;
          sred[2048 + b * 16 + lc] = acc[0][i] + bmu[cb + lc];
        }
      }
      __syncthreads();
      // elementwise mu = (1-gate)*xmu + gate*pm; stash relu(pm) for lv GEMM
      for (int i = threadIdx.x; i < 1024; i += blockDim.x) {
        int b = i >> 4, x = cb + (i & 15);
        float gate = sred[i], pm = sred[1024 + i], xmu = sred[2048 + i];
        float mu = (1.0f - gate) * xmu + gate * pm;
        out[PMU + (size_t)b * (Tn * Xn) + t * Xn + x] = mu;
        pmrelu[b * Xn + x] = fmaxf(pm, 0.0f);
        sred[i] = mu;
      }
      grid_barrier(cnt, gen, 8);
      // stage4: lv = relu(pm)@Wlv + blv (needs full pm -> after barrier)
      Ws[0] = WlvT;
      mm64(pmrelu, 0, Xn, Xn, Ws, (wave < 4) ? 1 : 0, mts, nbs, acc, smem);
      if (wave < 4) {
#pragma unroll
        for (int i = 0; i < 8; ++i) {
          int b = mts[0] * 16 + i + hi;
          float lv = acc[0][i] + blv[cb + lc];
          out[PLV + (size_t)b * (Tn * Xn) + t * Xn + cb + lc] = lv;
          sred[1024 + b * 16 + lc] = lv;
        }
      }
      __syncthreads();
      for (int i = threadIdx.x; i < 1024; i += blockDim.x) {
        int b = i >> 4, x = cb + (i & 15);
        float xt = sred[i] + eps_p[((size_t)t * Bn + b) * Xn + x] * expf(0.5f * sred[1024 + i]);
        out[PX + (size_t)b * (Tn * Xn) + t * Xn + x] = xt;
        xg_buf[((t + 1) & 1) * (Bn * Xn) + b * Xn + x] = xt;
      }
      grid_barrier(cnt, gen, 8);
    }
  }
}

// ---------------------------------------------------------------------------
extern "C" void kernel_launch(void* const* d_in, const int* in_sizes, int n_in,
                              void* d_out, int out_size, void* d_ws, size_t ws_size,
                              hipStream_t stream) {
  (void)in_sizes; (void)n_in; (void)out_size; (void)ws_size;
  const float* data = (const float*)d_in[0];
  const float* eps_q = (const float*)d_in[1];
  const float* eps_p = (const float*)d_in[2];
  const float* Wih = (const float*)d_in[3];
  const float* Whh = (const float*)d_in[4];
  const float* bih = (const float*)d_in[5];
  const float* bhh = (const float*)d_in[6];
  const float* h0  = (const float*)d_in[7];
  const float* xq0 = (const float*)d_in[8];
  const float* x0  = (const float*)d_in[9];
  const float* Wxh = (const float*)d_in[10];
  const float* bxh = (const float*)d_in[11];
  const float* Whm = (const float*)d_in[12];
  const float* bhm = (const float*)d_in[13];
  const float* Whl = (const float*)d_in[14];
  const float* bhl = (const float*)d_in[15];
  const float* Wg1 = (const float*)d_in[16];
  const float* bg1 = (const float*)d_in[17];
  const float* Wg2 = (const float*)d_in[18];
  const float* bg2 = (const float*)d_in[19];
  const float* Wp1 = (const float*)d_in[20];
  const float* bp1 = (const float*)d_in[21];
  const float* Wp2 = (const float*)d_in[22];
  const float* bp2 = (const float*)d_in[23];
  const float* Wmu = (const float*)d_in[24];
  const float* bmu = (const float*)d_in[25];
  const float* Wlv = (const float*)d_in[26];
  const float* blv = (const float*)d_in[27];
  const float* eW1 = (const float*)d_in[28];
  const float* eb1 = (const float*)d_in[29];
  const float* eW2 = (const float*)d_in[30];
  const float* eb2 = (const float*)d_in[31];
  const float* eW3 = (const float*)d_in[32];
  const float* eb3 = (const float*)d_in[33];

  // Workspace layout (bytes):
  //  [0, 128MB)        x_proj fp32 (T,B,R)      -> reused as h1 bf16 (B*T,E)
  //  [128MB, 192MB)    rnn_out bf16 (T,B,R)     -> reused as h2 bf16 (B*T,E)
  //  [192MB, ...)      small state + barriers + bf16 transposed weights
  const size_t XPROJ_BYTES = (size_t)Tn * Bn * Rn * 4;          // 134217728
  const size_t RNNOUT_BYTES = (size_t)Tn * Bn * Rn * 2;         // 67108864
  char* ws = (char*)d_ws;
  float*          xproj  = (float*)(ws);
  unsigned short* rnnout = (unsigned short*)(ws + XPROJ_BYTES);
  unsigned short* h1     = (unsigned short*)(ws);               // reuse
  unsigned short* h2     = (unsigned short*)(ws + XPROJ_BYTES); // reuse
  char* sm = ws + XPROJ_BYTES + RNNOUT_BYTES;
  float* h_buf  = (float*)(sm);                    // 2 * 64*512
  float* hc_buf = (float*)(sm + 262144);           // 64*512
  float* xq_buf = (float*)(sm + 393216);           // 2 * 64*128
  float* xg_buf = (float*)(sm + 458752);           // 2 * 64*128
  float* g1_buf = (float*)(sm + 524288);           // 64*512
  float* p1_buf = (float*)(sm + 655360);           // 64*512
  float* pmrelu = (float*)(sm + 786432);           // 64*128
  unsigned* bar = (unsigned*)(sm + 819200);        // 4 counters

  // bf16 transposed weights (N x K each), base 1MB into small region
  unsigned short* wb = (unsigned short*)(sm + 1048576);
  unsigned short* WihT = wb;             // 512x512
  unsigned short* WhhT = WihT + 262144;  // 512x512
  unsigned short* WxhT = WhhT + 262144;  // 512x128
  unsigned short* WhmT = WxhT + 65536;   // 128x512
  unsigned short* WhlT = WhmT + 65536;   // 128x512
  unsigned short* Wg1T = WhlT + 65536;   // 512x128
  unsigned short* Wg2T = Wg1T + 65536;   // 128x512
  unsigned short* Wp1T = Wg2T + 65536;   // 512x128
  unsigned short* Wp2T = Wp1T + 65536;   // 128x512
  unsigned short* WmuT = Wp2T + 65536;   // 128x128
  unsigned short* WlvT = WmuT + 16384;   // 128x128
  unsigned short* eW1T = WlvT + 16384;   // 512x128
  unsigned short* eW2T = eW1T + 65536;   // 512x512
  unsigned short* eW3T = eW2T + 262144;  // 512x512

  float* out = (float*)d_out;

  // --- one-time weight transpose/convert to bf16 (3 MB total, L2 resident) ---
  #define WT(W, D, K, N, LG) \
    wt_kernel<<<((K) * (N) + 255) / 256, 256, 0, stream>>>(W, D, (K) * (N), LG, K)
  WT(Wih, WihT, 512, 512, 9);
  WT(Whh, WhhT, 512, 512, 9);
  WT(Wxh, WxhT, 128, 512, 9);
  WT(Whm, WhmT, 512, 128, 7);
  WT(Whl, WhlT, 512, 128, 7);
  WT(Wg1, Wg1T, 128, 512, 9);
  WT(Wg2, Wg2T, 512, 128, 7);
  WT(Wp1, Wp1T, 128, 512, 9);
  WT(Wp2, Wp2T, 512, 128, 7);
  WT(Wmu, WmuT, 128, 128, 7);
  WT(Wlv, WlvT, 128, 128, 7);
  WT(eW1, eW1T, 128, 512, 9);
  WT(eW2, eW2T, 512, 512, 9);
  WT(eW3, eW3T, 512, 512, 9);
  #undef WT

  init_kernel<<<1, 256, 0, stream>>>(h_buf, xq_buf, xg_buf, bar, h0, xq0, x0);
  xproj_kernel<<<dim3(Tn, Rn / 64), 256, 0, stream>>>(data, WihT, bih, bhh, xproj);
  scan_kernel<<<16, 256, 0, stream>>>(xproj, WhhT, rnnout, WxhT, bxh, WhmT, bhm, WhlT, bhl,
                                      eps_q, Wg1T, bg1, Wg2T, bg2, Wp1T, bp1, Wp2T, bp2,
                                      WmuT, bmu, WlvT, blv, eps_p, out,
                                      h_buf, hc_buf, xq_buf, xg_buf, g1_buf, p1_buf,
                                      pmrelu, bar);
  // Emission MLP on q_x: h1 = relu(qx@eW1+b1); h2 = relu(h1@eW2+b2); y = sigmoid(h2@eW3+b3)
  gemm_act<<<dim3((Bn * Tn) / 64, En / 64), 256, 0, stream>>>(out + QX, 0, eW1T, eb1, h1, 1, Xn, En, 1);
  gemm_act<<<dim3((Bn * Tn) / 64, En / 64), 256, 0, stream>>>(h1, 1, eW2T, eb2, h2, 1, En, En, 1);
  gemm_act<<<dim3((Bn * Tn) / 64, Yn / 64), 256, 0, stream>>>(h2, 1, eW3T, eb3, out + YP, 0, En, Yn, 2);
}
